// ChunkedMHA_66812511257199
// MI455X (gfx1250) — matrix operand order repaired and checked by reading, not compiled, not run on hardware
//
#include <hip/hip_runtime.h>
#include <type_traits>

// ---------------------------------------------------------------------------
// Chunked sigmoid MHA for gfx1250 (MI455X), bf16 WMMA pipeline.
// out = ( sigmoid( (X Wq^T * s) (X Wk^T)^T ) (X Wv^T) ) Wo^T   per head.
// ---------------------------------------------------------------------------

typedef __attribute__((ext_vector_type(16))) __bf16 v16bf;
typedef __attribute__((ext_vector_type(8)))  __bf16 v8bf;
typedef __attribute__((ext_vector_type(8)))  float  v8f;

constexpr int E_ = 1024;   // embed dim
constexpr int H_ = 16;     // heads
constexpr int D_ = 64;     // head dim
constexpr int B_ = 2;
constexpr int L_ = 2048;   // query len
constexpr int S_ = 2048;   // kv len

// GEMM tiling
constexpr int BM = 128, BN = 64, BK = 32;
constexpr int LDT = 48;    // LDS row stride (bf16): 96B rows -> 32B-aligned frags

__device__ __forceinline__ v8f zero8() {
  v8f z;
#pragma unroll
  for (int i = 0; i < 8; ++i) z[i] = 0.0f;
  return z;
}

__device__ __forceinline__ v16bf combine16(v8bf lo, v8bf hi) {
  v16bf r;
#pragma unroll
  for (int i = 0; i < 8; ++i) { r[i] = lo[i]; r[i + 8] = hi[i]; }
  return r;
}

__device__ __forceinline__ v8bf cvt8(float4 a, float4 b) {
  v8bf r;
  r[0] = (__bf16)a.x; r[1] = (__bf16)a.y; r[2] = (__bf16)a.z; r[3] = (__bf16)a.w;
  r[4] = (__bf16)b.x; r[5] = (__bf16)b.y; r[6] = (__bf16)b.z; r[7] = (__bf16)b.w;
  return r;
}

__device__ __forceinline__ v8f wmma_bf16(v16bf a, v16bf b, v8f c) {
  // D = A(16x32 bf16) * B(32x16 bf16) + C(16x16 f32)
  return __builtin_amdgcn_wmma_f32_16x16x32_bf16(
      /*neg_a=*/false, a, /*neg_b=*/false, b,
      /*c_mod=*/(short)0, c, /*reuse_a=*/false, /*reuse_b=*/false);
}

// Fast sigmoid: v_exp_f32 + v_add + v_rcp_f32 (no IEEE division ladder);
// plenty accurate for a bf16-bound score in [0,1].
__device__ __forceinline__ float fast_sigmoid(float x) {
  return __builtin_amdgcn_rcpf(1.0f + __expf(-x));
}

// Output scatter: LAYOUT 0 -> [B,H,S,D] (q,k), 1 -> [B,H,D,S] (v transposed),
//                 2 -> [M,N] row-major (o intermediate / final out)
template <int LAYOUT, typename OutT>
__device__ __forceinline__ void storeC(OutT* out, int m, int n, float v) {
  if constexpr (LAYOUT == 2) {
    out[(size_t)m * E_ + n] = (OutT)v;
  } else {
    const int b = m >> 11, s = m & (L_ - 1);
    const int h = n >> 6,  d = n & (D_ - 1);
    if constexpr (LAYOUT == 0)
      out[(((size_t)(b * H_ + h) * S_) + s) * D_ + d] = (OutT)v;
    else
      out[(((size_t)(b * H_ + h) * D_) + d) * S_ + s] = (OutT)v;
  }
}

// C[M,N] = A[M,K] * W[N,K]^T  (A fp32 or bf16; W fp32; converted to bf16 in LDS)
template <typename AT, typename OutT, int LAYOUT>
__global__ void __launch_bounds__(256)
gemm_wmma(const AT* __restrict__ A, const float* __restrict__ W,
          OutT* __restrict__ out, int K, float scale) {
  __shared__ alignas(32) __bf16 As[BM * LDT];
  __shared__ alignas(32) __bf16 Ws[BN * LDT];

  const int tid  = threadIdx.x;
  const int lane = tid & 31;
  const int wid  = tid >> 5;
  const int blockN = blockIdx.x * BN;
  const int blockM = blockIdx.y * BM;
  const int mBase  = (wid & 3) * 32;   // 4 waves along M
  const int nBase  = (wid >> 2) * 32;  // 2 waves along N

  v8f acc[2][2];
#pragma unroll
  for (int i = 0; i < 2; ++i)
#pragma unroll
    for (int j = 0; j < 2; ++j) acc[i][j] = zero8();

  const int arow = tid >> 1, acb = (tid & 1) * 16;  // A stage: 128 rows x 32 k
  const int wrow = tid >> 2, wcb = (tid & 3) * 8;   // W stage:  64 rows x 32 k

  const int r0 = lane & 15;
  const int cA = (lane < 16) ? 0 : 16;   // A-frag K chunk base (contiguous K map)
  const int cB = (lane < 16) ? 0 : 16;   // B-frag K chunk base

  for (int k0 = 0; k0 < K; k0 += BK) {
    // ---- stage A tile (fp32 -> bf16, or bf16 passthrough) ----
    if constexpr (std::is_same<AT, float>::value) {
      const float* ap = A + (size_t)(blockM + arow) * K + k0 + acb;
      float4 f0 = ((const float4*)ap)[0];
      float4 f1 = ((const float4*)ap)[1];
      float4 f2 = ((const float4*)ap)[2];
      float4 f3 = ((const float4*)ap)[3];
      *(v8bf*)&As[arow * LDT + acb]     = cvt8(f0, f1);
      *(v8bf*)&As[arow * LDT + acb + 8] = cvt8(f2, f3);
    } else {
      const __bf16* ap = A + (size_t)(blockM + arow) * K + k0 + acb;
      *(v8bf*)&As[arow * LDT + acb]     = *(const v8bf*)ap;
      *(v8bf*)&As[arow * LDT + acb + 8] = *(const v8bf*)(ap + 8);
    }
    // ---- stage W tile ----
    {
      const float* wp = W + (size_t)(blockN + wrow) * K + k0 + wcb;
      float4 g0 = ((const float4*)wp)[0];
      float4 g1 = ((const float4*)wp)[1];
      *(v8bf*)&Ws[wrow * LDT + wcb] = cvt8(g0, g1);
    }
    __syncthreads();

    // ---- fragments: issue ALL ds loads, fence, then the 4 WMMAs ----
    v16bf af[2], bfr[2];
#pragma unroll
    for (int i = 0; i < 2; ++i) {
      const __bf16* p = &As[(mBase + 16 * i + r0) * LDT + cA];
      af[i] = combine16(*(const v8bf*)p, *(const v8bf*)(p + 8));
    }
#pragma unroll
    for (int j = 0; j < 2; ++j)
      bfr[j] = *(const v16bf*)&Ws[(nBase + 16 * j + r0) * LDT + cB];
    __builtin_amdgcn_sched_barrier(0);  // keep loads grouped ahead of WMMAs
#pragma unroll
    for (int i = 0; i < 2; ++i)
#pragma unroll
      for (int j = 0; j < 2; ++j)
        acc[i][j] = wmma_bf16(af[i], bfr[j], acc[i][j]);
    __syncthreads();
  }

  // ---- epilogue (C layout: vgpr r -> M=r(+8 for hi lanes), N=lane%16) ----
  const int mOff = (lane < 16) ? 0 : 8;
#pragma unroll
  for (int i = 0; i < 2; ++i)
#pragma unroll
    for (int j = 0; j < 2; ++j)
#pragma unroll
      for (int r = 0; r < 8; ++r) {
        const int m = blockM + mBase + 16 * i + r + mOff;
        const int n = blockN + nBase + 16 * j + r0;
        storeC<LAYOUT>(out, m, n, acc[i][j][r] * scale);
      }
}

// Fused sigmoid attention: one wave per 16 query rows; loops over 32-key chunks.
// q,k: [B,H,S,D] bf16 (q pre-scaled); vt: [B,H,D,S] bf16; o: [B,L,E] bf16.
__global__ void __launch_bounds__(256)
attn_sigmoid(const __bf16* __restrict__ q, const __bf16* __restrict__ k,
             const __bf16* __restrict__ vt, __bf16* __restrict__ o) {
  __shared__ alignas(32) __bf16 Ps[8 * 16 * 32];  // 1KB per wave relayout scratch

  const int tid = threadIdx.x, lane = tid & 31, wid = tid >> 5;
  const int g = blockIdx.x * 8 + wid;        // global wave id
  const int wavesPerBH = L_ / 16;            // 128
  const int bh    = g / wavesPerBH;
  const int qRow0 = (g % wavesPerBH) * 16;
  const int b = bh / H_, h = bh % H_;

  const __bf16* qp = q  + (size_t)bh * S_ * D_;
  const __bf16* kp = k  + (size_t)bh * S_ * D_;
  const __bf16* vp = vt + (size_t)bh * D_ * S_;
  __bf16* lds = &Ps[wid * 16 * 32];

  const int r0   = lane & 15;
  const int cA   = (lane < 16) ? 0 : 16;
  const int cB   = (lane < 16) ? 0 : 16;
  const int mOff = (lane < 16) ? 0 : 8;

  // Q A-fragments for d = 0..31 and 32..63, kept in registers.
  v16bf qf[2];
#pragma unroll
  for (int dstep = 0; dstep < 2; ++dstep) {
    const __bf16* p = qp + (size_t)(qRow0 + r0) * D_ + dstep * 32 + cA;
    qf[dstep] = combine16(*(const v8bf*)p, *(const v8bf*)(p + 8));
  }

  v8f oacc[4];
#pragma unroll
  for (int j = 0; j < 4; ++j) oacc[j] = zero8();

  for (int kc = 0; kc < S_; kc += 32) {
    // ---- issue ALL K and V fragment loads up front (16x b128 in flight) ----
    v16bf kf[2][2];   // [dstep][key-tile]
#pragma unroll
    for (int dstep = 0; dstep < 2; ++dstep)
#pragma unroll
      for (int j = 0; j < 2; ++j)
        kf[dstep][j] = *(const v16bf*)
            (kp + (size_t)(kc + 16 * j + r0) * D_ + dstep * 32 + cB);
    v16bf vf[4];
#pragma unroll
    for (int j2 = 0; j2 < 4; ++j2)
      vf[j2] = *(const v16bf*)(vp + (size_t)(16 * j2 + r0) * S_ + kc + cB);
    if (kc + 32 < S_) {  // hint next K/V chunk toward the caches
      __builtin_prefetch(kp + (size_t)(kc + 32) * D_, 0, 1);
      __builtin_prefetch(vp + kc + 32, 0, 1);
    }
    __builtin_amdgcn_sched_barrier(0);  // loads stay grouped above the WMMAs

    // ---- P = Q * K^T (4 WMMAs) ----
    v8f pacc[2] = {zero8(), zero8()};
#pragma unroll
    for (int dstep = 0; dstep < 2; ++dstep)
#pragma unroll
      for (int j = 0; j < 2; ++j)
        pacc[j] = wmma_bf16(qf[dstep], kf[dstep][j], pacc[j]);
    __builtin_amdgcn_sched_barrier(0);

    // ---- sigmoid (fast rcp), C-layout -> A-layout via per-wave LDS bounce ----
#pragma unroll
    for (int j = 0; j < 2; ++j)
#pragma unroll
      for (int r = 0; r < 8; ++r)
        lds[(r + mOff) * 32 + 16 * j + r0] = (__bf16)fast_sigmoid(pacc[j][r]);
    // DS ops are in-order within a wave; fence the compiler + drain dscnt.
    asm volatile("s_wait_dscnt 0" ::: "memory");
    const __bf16* pp = &lds[r0 * 32 + cA];
    v16bf pf = combine16(*(const v8bf*)pp, *(const v8bf*)(pp + 8));

    // ---- O += P * V (4 WMMAs, V fragments already resident) ----
#pragma unroll
    for (int j2 = 0; j2 < 4; ++j2)
      oacc[j2] = wmma_bf16(pf, vf[j2], oacc[j2]);
  }

  // ---- epilogue: o[b, s, h*64 + d] bf16 ----
#pragma unroll
  for (int j2 = 0; j2 < 4; ++j2)
#pragma unroll
    for (int r = 0; r < 8; ++r) {
      const int s   = qRow0 + r + mOff;
      const int col = h * 64 + 16 * j2 + r0;
      o[((size_t)b * L_ + s) * E_ + col] = (__bf16)oacc[j2][r];
    }
}

extern "C" void kernel_launch(void* const* d_in, const int* in_sizes, int n_in,
                              void* d_out, int out_size, void* d_ws, size_t ws_size,
                              hipStream_t stream) {
  (void)in_sizes; (void)n_in; (void)out_size; (void)ws_size;
  const float* Xq = (const float*)d_in[0];
  const float* Xk = (const float*)d_in[1];
  const float* Xv = (const float*)d_in[2];
  const float* Wq = (const float*)d_in[3];
  const float* Wk = (const float*)d_in[4];
  const float* Wv = (const float*)d_in[5];
  const float* Wo = (const float*)d_in[6];
  float* out = (float*)d_out;

  const size_t plane = (size_t)B_ * H_ * S_ * D_;  // 4M elems (8MB bf16 each)
  __bf16* qb = (__bf16*)d_ws;
  __bf16* kb = qb + plane;
  __bf16* vb = kb + plane;
  __bf16* ob = vb + plane;  // [B, L, E] bf16 attention output

  const int M = B_ * L_, N = E_, K = E_;
  dim3 gg(N / BN, M / BM);  // 16 x 32 blocks, 256 threads (8 waves) each
  const float scale = 0.125f;  // D^-0.5

  gemm_wmma<float,  __bf16, 0><<<gg, 256, 0, stream>>>(Xq, Wq, qb, K, scale);
  gemm_wmma<float,  __bf16, 0><<<gg, 256, 0, stream>>>(Xk, Wk, kb, K, 1.0f);
  gemm_wmma<float,  __bf16, 1><<<gg, 256, 0, stream>>>(Xv, Wv, vb, K, 1.0f);

  attn_sigmoid<<<(B_ * H_ * (L_ / 16)) / 8, 256, 0, stream>>>(qb, kb, vb, ob);

  gemm_wmma<__bf16, float,  2><<<gg, 256, 0, stream>>>(ob, Wo, out, K, 1.0f);
}
